// VisionSdpaAttention_2499670966342
// MI455X (gfx1250) — matrix operand order repaired and checked by reading, not compiled, not run on hardware
//
#include <hip/hip_runtime.h>
#include <hip/hip_bf16.h>
#include <math.h>
#include <stdint.h>

typedef _Float16 f16;
typedef __attribute__((ext_vector_type(16))) _Float16 v16h;
typedef __attribute__((ext_vector_type(8)))  _Float16 h8;
typedef __attribute__((ext_vector_type(8)))  float    v8f;

#define SHUF16(lo,hi) __builtin_shufflevector((lo),(hi),0,1,2,3,4,5,6,7,8,9,10,11,12,13,14,15)
#define WMMA_F16(a,b,c) __builtin_amdgcn_wmma_f32_16x16x32_f16(false,(a),false,(b),(short)0,(c),false,false)

static constexpr int S_LEN = 2048;
static constexpr int BATCH = 4;
static constexpr int DM    = 1280;   // model dim
static constexpr int NH    = 16;
static constexpr int DH    = 80;     // head dim
static constexpr int DHP   = 96;     // padded head dim (3 x 32 for WMMA K)
static constexpr int NTOK  = S_LEN * BATCH;          // 8192
static constexpr size_t QKSTRIDE = (size_t)BATCH * NH * S_LEN * DHP; // halves per q/k/v

// ---------------------------------------------------------------------------
// CDNA5 async copy: global -> LDS, 16B per lane, tracked by ASYNCcnt.
// Low 32 bits of a generic LDS pointer are the LDS byte offset (the LDS
// aperture check uses addr[63:32] only).
// ---------------------------------------------------------------------------
__device__ __forceinline__ void async_copy16(const f16* lds_dst, const f16* gsrc) {
  unsigned ldsOff = (unsigned)(uintptr_t)lds_dst;
  unsigned long long ga = (unsigned long long)(uintptr_t)gsrc;
  asm volatile("global_load_async_to_lds_b128 %0, %1, off"
               :: "v"(ldsOff), "v"(ga) : "memory");
}
__device__ __forceinline__ void wait_async() {
  asm volatile("s_wait_asynccnt 0x0" ::: "memory");
}

// ---------------------------------------------------------------------------
// elementwise helpers
// ---------------------------------------------------------------------------
__global__ void cvt_kernel(const float* __restrict__ src, f16* __restrict__ dst, int n) {
  int i = blockIdx.x * blockDim.x + threadIdx.x;
  int stride = gridDim.x * blockDim.x;
  for (; i < n; i += stride) dst[i] = (f16)src[i];
}

__global__ void zero_kernel(f16* __restrict__ dst, int n) {
  int i = blockIdx.x * blockDim.x + threadIdx.x;
  int stride = gridDim.x * blockDim.x;
  for (; i < n; i += stride) dst[i] = (f16)0.0f;
}

// RoPE on q and k (in-place, f16 storage, fp32 math). Folds softmax scale into q.
__global__ void rope_kernel(f16* __restrict__ qkv, const float* __restrict__ freqs) {
  const float scale = 0.11180339887498949f;  // 1/sqrt(80)
  int n = 2 * BATCH * NH * S_LEN * 40;       // pairs (j, j+40), q and k
  int i = blockIdx.x * blockDim.x + threadIdx.x;
  int stride = gridDim.x * blockDim.x;
  for (; i < n; i += stride) {
    int f    = i % 40;
    int rest = i / 40;
    int s    = rest % S_LEN;
    int bh   = (rest / S_LEN) & 63;
    int wq   = i / (40 * S_LEN * 64);  // 0 = q, 1 = k
    float fr = freqs[s * 40 + f];
    float c  = __cosf(fr);
    float sn = __sinf(fr);
    size_t base = (size_t)wq * QKSTRIDE + ((size_t)bh * S_LEN + s) * DHP;
    float x0 = (float)qkv[base + f];
    float x1 = (float)qkv[base + f + 40];
    float y0 = x0 * c - x1 * sn;
    float y1 = x1 * c + x0 * sn;
    if (wq == 0) { y0 *= scale; y1 *= scale; }
    qkv[base + f]      = (f16)y0;
    qkv[base + f + 40] = (f16)y1;
  }
}

// ---------------------------------------------------------------------------
// Shared WMMA GEMM mainloop: block tile 128M x 128N, K step 32, 8 waves.
// Wave w: mw = w&3 -> M strip of 32 (2 x 16 sub-tiles); nw = w>>2 -> N strip
// of 64 (4 x 16 tiles). 8 WMMAs per K-step per wave, B fragments reused
// across both M sub-tiles. Tiles staged with async global->LDS copies.
// ---------------------------------------------------------------------------
__device__ __forceinline__ void gemm_mainloop(
    const f16* __restrict__ A, const f16* __restrict__ Bm,
    int m0, int n0, int K, v8f acc[2][4],
    f16 (*As)[40], f16 (*Bs)[40])
{
  const int tid  = threadIdx.x;
  const int lane = tid & 31;
  const int w    = tid >> 5;
  const int mw   = w & 3;
  const int nw   = w >> 2;
  const int col  = lane & 15;
  const int kbA  = (lane < 16) ? 0 : 8;    // A: K 0-7/16-23 | 8-15/24-31
  const int kbB  = (lane < 16) ? 0 : 8;    // B: K 0-7/16-23 | 8-15/24-31

  for (int kk = 0; kk < K; kk += 32) {
    __syncthreads();
    // stage A and B: 128 rows x 32 halves each; 256 threads x (16B A + 16B B) x 2
#pragma unroll
    for (int hh = 0; hh < 2; ++hh) {
      int c = tid + hh * 256;              // 0..511 chunks of 8 halves
      int row = c >> 2, seg = c & 3;
      async_copy16(&As[row][seg * 8], A  + (size_t)(m0 + row) * K + kk + seg * 8);
      async_copy16(&Bs[row][seg * 8], Bm + (size_t)(n0 + row) * K + kk + seg * 8);
    }
    wait_async();
    __syncthreads();
    v16h a[2];
#pragma unroll
    for (int mi = 0; mi < 2; ++mi) {
      const f16* ap = &As[mw * 32 + mi * 16 + col][kbA];
      h8 alo = *(const h8*)ap;
      h8 ahi = *(const h8*)(ap + 16);
      a[mi] = SHUF16(alo, ahi);
    }
#pragma unroll
    for (int j = 0; j < 4; ++j) {
      const f16* bp = &Bs[nw * 64 + j * 16 + col][kbB];
      h8 blo = *(const h8*)bp;
      h8 bhi = *(const h8*)(bp + 16);
      v16h b = SHUF16(blo, bhi);
      acc[0][j] = WMMA_F16(a[0], b, acc[0][j]);
      acc[1][j] = WMMA_F16(a[1], b, acc[1][j]);
    }
  }
}

// GEMM 1: qkv[t][e] = X[t][:] . Wqkv[e][:] + bias[e], scattered to [3][B][H][S][96] f16
__global__ void __launch_bounds__(256) qkv_gemm_kernel(
    const f16* __restrict__ X, const f16* __restrict__ W,
    const float* __restrict__ bias, f16* __restrict__ qkv)
{
  __shared__ alignas(16) f16 As[128][40];
  __shared__ alignas(16) f16 Bs[128][40];
  int m0 = blockIdx.y * 128, n0 = blockIdx.x * 128;
  v8f acc[2][4];
#pragma unroll
  for (int mi = 0; mi < 2; ++mi)
#pragma unroll
    for (int j = 0; j < 4; ++j)
#pragma unroll
      for (int i = 0; i < 8; ++i) acc[mi][j][i] = 0.0f;

  gemm_mainloop(X, W, m0, n0, DM, acc, As, Bs);

  const int lane = threadIdx.x & 31;
  const int w = threadIdx.x >> 5;
  const int mw = w & 3, nw = w >> 2;
  const int hi = lane >> 4;
#pragma unroll
  for (int mi = 0; mi < 2; ++mi) {
    int rbase = m0 + mw * 32 + mi * 16 + hi * 8;  // C layout: vgpr i -> row i (+8 hi lanes)
#pragma unroll
    for (int j = 0; j < 4; ++j) {
      int e = n0 + nw * 64 + j * 16 + (lane & 15);
      int which = e / DM;
      int r = e - which * DM;
      int h = r / DH;
      int dh = r - h * DH;
      float bv = bias[e];
#pragma unroll
      for (int i = 0; i < 8; ++i) {
        int t = rbase + i;
        int s = t >> 2;    // t = s*BATCH + b
        int b = t & 3;
        qkv[(size_t)which * QKSTRIDE + ((size_t)(b * NH + h) * S_LEN + s) * DHP + dh] =
            (f16)(acc[mi][j][i] + bv);
      }
    }
  }
}

// GEMM 2: out[t][e] = Aout[t][:] . Wout[e][:] + bias[e], fp32 output
__global__ void __launch_bounds__(256) out_gemm_kernel(
    const f16* __restrict__ Aout, const f16* __restrict__ W,
    const float* __restrict__ bias, float* __restrict__ out)
{
  __shared__ alignas(16) f16 As[128][40];
  __shared__ alignas(16) f16 Bs[128][40];
  int m0 = blockIdx.y * 128, n0 = blockIdx.x * 128;
  v8f acc[2][4];
#pragma unroll
  for (int mi = 0; mi < 2; ++mi)
#pragma unroll
    for (int j = 0; j < 4; ++j)
#pragma unroll
      for (int i = 0; i < 8; ++i) acc[mi][j][i] = 0.0f;

  gemm_mainloop(Aout, W, m0, n0, DM, acc, As, Bs);

  const int lane = threadIdx.x & 31;
  const int w = threadIdx.x >> 5;
  const int mw = w & 3, nw = w >> 2;
  const int hi = lane >> 4;
#pragma unroll
  for (int mi = 0; mi < 2; ++mi) {
    int rbase = m0 + mw * 32 + mi * 16 + hi * 8;
#pragma unroll
    for (int j = 0; j < 4; ++j) {
      int e = n0 + nw * 64 + j * 16 + (lane & 15);
      float bv = bias[e];
#pragma unroll
      for (int i = 0; i < 8; ++i) {
        int t = rbase + i;
        out[(size_t)t * DM + e] = acc[mi][j][i] + bv;
      }
    }
  }
}

// ---------------------------------------------------------------------------
// Flash attention: grid (S/64, B*H), block 128 = 4 waves; each wave owns a
// 16-row q tile, streams 32-key tiles through LDS with online softmax.
// q already carries the 1/sqrt(Dh) scale (folded in RoPE).
// ---------------------------------------------------------------------------
__global__ void __launch_bounds__(128) attn_kernel(
    const f16* __restrict__ qkv, f16* __restrict__ aout)
{
  __shared__ alignas(16) f16 Ks[32][104];     // [key][d], padded stride
  __shared__ alignas(16) f16 Vt[DHP][40];     // [dh][key]
  __shared__ alignas(16) f16 Ps[4][16][40];   // per-wave P transpose buffer

  const int tid  = threadIdx.x;
  const int lane = tid & 31;
  const int w    = tid >> 5;
  const int bh   = blockIdx.y;                 // b*NH + h
  const int qbase = blockIdx.x * 64 + w * 16;
  const int col  = lane & 15;
  const int hi   = lane >> 4;
  const int kbA  = hi ? 8 : 0;
  const int kbB  = hi ? 8 : 0;

  const f16* qp = qkv;
  const f16* kp = qkv + QKSTRIDE;
  const f16* vp = qkv + 2 * QKSTRIDE;

  // Q fragments: A layout, rows qbase..qbase+15, d in 3 chunks of 32 (zero-padded 80..95)
  v16h qa[3];
  {
    const f16* qrow = qp + ((size_t)bh * S_LEN + qbase + col) * DHP;
#pragma unroll
    for (int i = 0; i < 3; ++i) {
      const f16* p = qrow + i * 32 + kbA;
      h8 lo = *(const h8*)p;
      h8 hh = *(const h8*)(p + 16);
      qa[i] = SHUF16(lo, hh);
    }
  }

  float m[8], l[8];
  v8f o[5];
#pragma unroll
  for (int i = 0; i < 8; ++i) { m[i] = -INFINITY; l[i] = 0.0f; }
#pragma unroll
  for (int j = 0; j < 5; ++j)
#pragma unroll
    for (int i = 0; i < 8; ++i) o[j][i] = 0.0f;

  for (int kt = 0; kt < S_LEN; kt += 32) {
    __syncthreads();
    // stage K tile via async copies; V tile loaded + transposed into LDS
#pragma unroll
    for (int i = 0; i < 3; ++i) {
      int c   = tid + i * 128;       // 0..383 chunks of 8 halves
      int row = c / 12;
      int off = (c % 12) * 8;
      async_copy16(&Ks[row][off], kp + ((size_t)bh * S_LEN + kt + row) * DHP + off);
      h8 vv = *(const h8*)(vp + ((size_t)bh * S_LEN + kt + row) * DHP + off);
#pragma unroll
      for (int jj = 0; jj < 8; ++jj) Vt[off + jj][row] = vv[jj];
    }
    wait_async();
    __syncthreads();

    // scores: two 16-key tiles, K-dim = 96
    v8f c0, c1;
#pragma unroll
    for (int i = 0; i < 8; ++i) { c0[i] = 0.0f; c1[i] = 0.0f; }
#pragma unroll
    for (int i = 0; i < 3; ++i) {
      {
        const f16* p = &Ks[col][i * 32 + kbB];
        h8 lo = *(const h8*)p; h8 hh = *(const h8*)(p + 16);
        v16h b = SHUF16(lo, hh);
        c0 = WMMA_F16(qa[i], b, c0);
      }
      {
        const f16* p = &Ks[16 + col][i * 32 + kbB];
        h8 lo = *(const h8*)p; h8 hh = *(const h8*)(p + 16);
        v16h b = SHUF16(lo, hh);
        c1 = WMMA_F16(qa[i], b, c1);
      }
    }

    // online softmax (per-row stats; rows live in vgpr index, keys across 16 lanes)
    float rm[8];
#pragma unroll
    for (int i = 0; i < 8; ++i) rm[i] = fmaxf(c0[i], c1[i]);
#pragma unroll
    for (int d = 1; d < 16; d <<= 1)
#pragma unroll
      for (int i = 0; i < 8; ++i) rm[i] = fmaxf(rm[i], __shfl_xor(rm[i], d, 32));

    float al[8], rs[8];
    v8f p0, p1;
#pragma unroll
    for (int i = 0; i < 8; ++i) {
      float mn = fmaxf(m[i], rm[i]);
      al[i] = __expf(m[i] - mn);
      m[i] = mn;
      p0[i] = __expf(c0[i] - mn);
      p1[i] = __expf(c1[i] - mn);
      rs[i] = p0[i] + p1[i];
    }
#pragma unroll
    for (int d = 1; d < 16; d <<= 1)
#pragma unroll
      for (int i = 0; i < 8; ++i) rs[i] += __shfl_xor(rs[i], d, 32);
#pragma unroll
    for (int i = 0; i < 8; ++i) l[i] = l[i] * al[i] + rs[i];
#pragma unroll
    for (int j = 0; j < 5; ++j)
#pragma unroll
      for (int i = 0; i < 8; ++i) o[j][i] *= al[i];

    // transpose P (C layout -> A layout) via per-wave LDS
#pragma unroll
    for (int i = 0; i < 8; ++i) {
      int M = hi * 8 + i;
      Ps[w][M][col]      = (f16)p0[i];
      Ps[w][M][16 + col] = (f16)p1[i];
    }
    asm volatile("s_wait_dscnt 0" ::: "memory");
    v16h pa;
    {
      const f16* p = &Ps[w][col][kbA];
      h8 lo = *(const h8*)p;
      h8 hh = *(const h8*)(p + 16);
      pa = SHUF16(lo, hh);
    }
    // o += P @ V : 5 dh tiles of 16 (covers dh 0..79)
#pragma unroll
    for (int j = 0; j < 5; ++j) {
      const f16* p = &Vt[j * 16 + col][kbB];
      h8 lo = *(const h8*)p; h8 hh = *(const h8*)(p + 16);
      v16h b = SHUF16(lo, hh);
      o[j] = WMMA_F16(pa, b, o[j]);
    }
  }

  // epilogue: aout[t][h*80+dh] = o / l   (t = s*BATCH + b)
  const int b = bh >> 4;
  const int h = bh & 15;
#pragma unroll
  for (int j = 0; j < 5; ++j) {
    int dh = j * 16 + col;
#pragma unroll
    for (int i = 0; i < 8; ++i) {
      int s = qbase + hi * 8 + i;
      int t = s * BATCH + b;
      aout[(size_t)t * DM + h * DH + dh] = (f16)(o[j][i] / l[i]);
    }
  }
}

// ---------------------------------------------------------------------------
extern "C" void kernel_launch(void* const* d_in, const int* in_sizes, int n_in,
                              void* d_out, int out_size, void* d_ws, size_t ws_size,
                              hipStream_t stream) {
  const float* hidden = (const float*)d_in[0];   // [S,B,D]
  const float* rotf   = (const float*)d_in[1];   // [S,40]
  const float* wqkv   = (const float*)d_in[2];   // [3D,D]
  const float* bqkv   = (const float*)d_in[3];   // [3D]
  const float* wout   = (const float*)d_in[4];   // [D,D]
  const float* bout   = (const float*)d_in[5];   // [D]
  float* out = (float*)d_out;

  char* ws = (char*)d_ws;
  size_t off = 0;
  f16* X16   = (f16*)(ws + off); off += (size_t)NTOK * DM * 2;        // 20.97 MB
  f16* Wq16  = (f16*)(ws + off); off += (size_t)3 * DM * DM * 2;      //  9.83 MB
  f16* Wo16  = (f16*)(ws + off); off += (size_t)DM * DM * 2;          //  3.28 MB
  f16* qkv16 = (f16*)(ws + off); off += 3 * QKSTRIDE * 2;             // 75.50 MB
  f16* ao16  = (f16*)(ws + off); off += (size_t)NTOK * DM * 2;        // 20.97 MB

  cvt_kernel<<<2048, 256, 0, stream>>>(hidden, X16, NTOK * DM);
  cvt_kernel<<<1024, 256, 0, stream>>>(wqkv, Wq16, 3 * DM * DM);
  cvt_kernel<<<512, 256, 0, stream>>>(wout, Wo16, DM * DM);
  zero_kernel<<<2048, 256, 0, stream>>>(qkv16, (int)(3 * QKSTRIDE));

  qkv_gemm_kernel<<<dim3(3 * DM / 128, NTOK / 128), 256, 0, stream>>>(X16, Wq16, bqkv, qkv16);
  rope_kernel<<<4096, 256, 0, stream>>>(qkv16, rotf);
  attn_kernel<<<dim3(S_LEN / 64, BATCH * NH), 128, 0, stream>>>(qkv16, ao16);
  out_gemm_kernel<<<dim3(DM / 128, NTOK / 128), 256, 0, stream>>>(ao16, Wo16, bout, out);
}
